// ShootingBlock2_86870008529682
// MI455X (gfx1250) — matrix operand order repaired and checked by reading, not compiled, not run on hardware
//
#include <hip/hip_runtime.h>

#define D_DIM 512
#define K_DIM 512
#define M_DIM 8192
#define T_DIM 10
#define N_SUBS 5

typedef __bf16 bf16_t;
typedef __attribute__((ext_vector_type(16))) bf16_t v16bf;
typedef __attribute__((ext_vector_type(8)))  float  v8f;

// ---------------------------------------------------------------------------
// bf16 WMMA GEMM:  C[m][n] = alpha * sum_k Aop[m][k]*Bop[k][n]  (+epilogue)
//   TA:      Aop[m][k] = A[k*512 + m]     (else A[m*512+k], float4 fast path)
//   RELU_A:  relu applied to A elements
//   BMODE:   0 = B is fp32, Bop[k][n] = B[k*512+n] (strided, optional RELU_B)
//            1 = B is packed bf16, Bop[k][n] = Bbf[n*512+k] (one v16bf load,
//                zero VALU; used with pre-packed theta^T / theta-transposed)
//   EPI: 0 = alpha*acc   1 = alpha*acc + bias[n]
//        2 = (Qmask[m][n] >= 0 ? alpha*acc : 0)
//   MT:  row tiles per wave (wave computes MT*16 rows x 64 cols,
//        each B fragment feeds MT WMMAs)
// Block = (32,8): 8 waves -> MT*128 rows x 64 cols per block.
// ---------------------------------------------------------------------------
template <bool TA, bool RELU_A, int BMODE, bool RELU_B, int EPI, int MT>
__global__ __launch_bounds__(256) void gemm_wmma_kernel(
    const float* __restrict__ A, const void* __restrict__ Bp,
    float* __restrict__ C, const float* __restrict__ bias,
    const float* __restrict__ Qmask, float alpha, int Mrows) {
  const int lane = threadIdx.x & 31;
  const int wave = threadIdx.y;
  const int l15  = lane & 15;
  const int lhi  = lane >> 4;
  const int rowBase = (blockIdx.y * 8 + wave) * (16 * MT);
  const int colBase = blockIdx.x * 64;
  if (rowBase >= Mrows) return;  // wave-uniform; grids are exact anyway

  v8f acc[MT][4];
#pragma unroll
  for (int mt = 0; mt < MT; ++mt)
#pragma unroll
    for (int t = 0; t < 4; ++t)
      acc[mt][t] = (v8f){0.f, 0.f, 0.f, 0.f, 0.f, 0.f, 0.f, 0.f};

  for (int kk = 0; kk < D_DIM; kk += 32) {
    // ---- A fragments: 16x32 bf16 each; lane l15 = row, lhi = K-half ----
    v16bf af[MT];
#pragma unroll
    for (int mt = 0; mt < MT; ++mt) {
      const int m = rowBase + mt * 16 + l15;
      if constexpr (!TA) {
        const int kb = kk + lhi * 16;  // lanes0-15: K+0..15; lanes16-31: +16..31
        const float4* pa = reinterpret_cast<const float4*>(A + (size_t)m * D_DIM + kb);
        __builtin_prefetch(pa + 8, 0, 1);  // next K-chunk of this row (speculative)
        float4 a0 = pa[0], a1 = pa[1], a2 = pa[2], a3 = pa[3];
        float av[16] = {a0.x, a0.y, a0.z, a0.w, a1.x, a1.y, a1.z, a1.w,
                        a2.x, a2.y, a2.z, a2.w, a3.x, a3.y, a3.z, a3.w};
#pragma unroll
        for (int j = 0; j < 16; ++j) {
          float v = av[j];
          if constexpr (RELU_A) v = fmaxf(v, 0.f);
          af[mt][j] = (bf16_t)v;
        }
      } else {
#pragma unroll
        for (int j = 0; j < 16; ++j) {
          const int kidx = kk + lhi * 16 + j;
          float v = A[(size_t)kidx * D_DIM + m];
          if constexpr (RELU_A) v = fmaxf(v, 0.f);
          af[mt][j] = (bf16_t)v;
        }
      }
    }
    // ---- 4 B fragments; each feeds MT WMMAs ----
#pragma unroll
    for (int t = 0; t < 4; ++t) {
      const int n  = colBase + t * 16 + l15;
      const int kb = kk + lhi * 16;  // lanes0-15: K=kk..kk+15; lanes16-31: +16..31
      v16bf bf;
      if constexpr (BMODE == 1) {
        bf = *reinterpret_cast<const v16bf*>(
            (const bf16_t*)Bp + (size_t)n * D_DIM + kb);
      } else {
        const float* Bf = (const float*)Bp;
#pragma unroll
        for (int j = 0; j < 16; ++j) {
          float v = Bf[(size_t)(kb + j) * D_DIM + n];
          if constexpr (RELU_B) v = fmaxf(v, 0.f);
          bf[j] = (bf16_t)v;
        }
      }
#pragma unroll
      for (int mt = 0; mt < MT; ++mt)
        acc[mt][t] = __builtin_amdgcn_wmma_f32_16x16x32_bf16(
            false, af[mt], false, bf, (short)0, acc[mt][t], false, false);
    }
  }

  // ---- epilogue: lane l15 = col n, acc element r -> row r + 8*lhi ----
#pragma unroll
  for (int mt = 0; mt < MT; ++mt)
#pragma unroll
    for (int t = 0; t < 4; ++t) {
      const int n = colBase + t * 16 + l15;
      float badd = 0.f;
      if constexpr (EPI == 1) badd = bias[n];
#pragma unroll
      for (int r = 0; r < 8; ++r) {
        const int m = rowBase + mt * 16 + r + lhi * 8;
        float v = acc[mt][t][r] * alpha + badd;
        if constexpr (EPI == 2) v = (Qmask[(size_t)m * D_DIM + n] >= 0.f) ? v : 0.f;
        C[(size_t)m * D_DIM + n] = v;
      }
    }
}

// Pack theta (512x512 fp32) into bf16:
//   tT[r*512+c] = bf16(theta[r][c])   (identity copy; B^T usage: dot_x/dot_q)
//   tC[n*512+k] = bf16(theta[k][n])   (LDS tile transpose;   usage: dot_p)
__global__ __launch_bounds__(256) void pack_theta_kernel(
    const float* __restrict__ theta, bf16_t* __restrict__ tT,
    bf16_t* __restrict__ tC) {
  __shared__ float tile[32][33];
  const int tx = threadIdx.x;      // 0..31
  const int ty = threadIdx.y;      // 0..7
  const int bx = blockIdx.x * 32;  // col (n) base
  const int by = blockIdx.y * 32;  // row (k) base
#pragma unroll
  for (int i = 0; i < 4; ++i) {
    const int r = ty + 8 * i;
    float v = theta[(size_t)(by + r) * D_DIM + bx + tx];
    tT[(size_t)(by + r) * D_DIM + bx + tx] = (bf16_t)v;
    tile[r][tx] = v;
  }
  __syncthreads();
#pragma unroll
  for (int i = 0; i < 4; ++i) {
    const int r = ty + 8 * i;  // output row n = bx + r
    tC[(size_t)(bx + r) * D_DIM + by + tx] = (bf16_t)tile[tx][r];
  }
}

// svec[d] = sum_k P[k][d]
__global__ void colsum_kernel(const float* __restrict__ P, float* __restrict__ s) {
  int d = blockIdx.x * blockDim.x + threadIdx.x;
  if (d >= D_DIM) return;
  float a = 0.f;
  for (int k = 0; k < K_DIM; ++k) a += P[(size_t)k * D_DIM + d];
  s[d] = a;
}

// b[i] = -sum_j Mbar_b[i][j] * s[j]
__global__ void bvec_kernel(const float* __restrict__ Mb, const float* __restrict__ s,
                            float* __restrict__ b) {
  int i = blockIdx.x * blockDim.x + threadIdx.x;
  if (i >= D_DIM) return;
  float a = 0.f;
  for (int j = 0; j < D_DIM; ++j) a += Mb[(size_t)i * D_DIM + j] * s[j];
  b[i] = -a;
}

// tmp = state + (cmul*dt)*k ;  acc = accInit ? k : acc + accw*k   (float4-wide)
__global__ void ew_stage_kernel(float4* __restrict__ tmp, float4* __restrict__ acc,
                                const float4* __restrict__ state,
                                const float4* __restrict__ k,
                                const float* __restrict__ bt, int iv, float cmul,
                                float accw, int accInit, int n4) {
  int i = blockIdx.x * blockDim.x + threadIdx.x;
  if (i >= n4) return;
  float dt = (bt[iv + 1] - bt[iv]) * (1.0f / (float)N_SUBS);
  float c  = cmul * dt;
  float4 kv = k[i], sv = state[i];
  float4 tv;
  tv.x = fmaf(c, kv.x, sv.x); tv.y = fmaf(c, kv.y, sv.y);
  tv.z = fmaf(c, kv.z, sv.z); tv.w = fmaf(c, kv.w, sv.w);
  tmp[i] = tv;
  if (accInit) {
    acc[i] = kv;
  } else {
    float4 av = acc[i];
    av.x = fmaf(accw, kv.x, av.x); av.y = fmaf(accw, kv.y, av.y);
    av.z = fmaf(accw, kv.z, av.z); av.w = fmaf(accw, kv.w, av.w);
    acc[i] = av;
  }
}

// state += (dt/6) * (acc + k4)   (float4-wide)
__global__ void ew_final_kernel(float4* __restrict__ state, const float4* __restrict__ acc,
                                const float4* __restrict__ k,
                                const float* __restrict__ bt, int iv, int n4) {
  int i = blockIdx.x * blockDim.x + threadIdx.x;
  if (i >= n4) return;
  float c = (bt[iv + 1] - bt[iv]) * (1.0f / (float)N_SUBS) * (1.0f / 6.0f);
  float4 kv = k[i], av = acc[i], sv = state[i];
  sv.x = fmaf(c, av.x + kv.x, sv.x); sv.y = fmaf(c, av.y + kv.y, sv.y);
  sv.z = fmaf(c, av.z + kv.z, sv.z); sv.w = fmaf(c, av.w + kv.w, sv.w);
  state[i] = sv;
}

extern "C" void kernel_launch(void* const* d_in, const int* in_sizes, int n_in,
                              void* d_out, int out_size, void* d_ws, size_t ws_size,
                              hipStream_t stream) {
  const float* x_in  = (const float*)d_in[0];  // (M,1,D)
  const float* bt    = (const float*)d_in[1];  // (T,)
  const float* q_in  = (const float*)d_in[2];  // (K,1,D)
  const float* p_in  = (const float*)d_in[3];  // (K,1,D)
  const float* Mbar  = (const float*)d_in[4];  // (D,D)
  const float* Mbarb = (const float*)d_in[5];  // (D,D)
  float* out = (float*)d_out;                  // (T,M,1,D)
  float* ws  = (float*)d_ws;

  const size_t SX = (size_t)M_DIM * D_DIM;
  const size_t SQ = (size_t)K_DIM * D_DIM;
  const size_t SN = SX + 2 * SQ;  // full state length (x|Q|P)

  float* state = ws;
  float* tmp   = ws + SN;
  float* kbuf  = ws + 2 * SN;
  float* accb  = ws + 3 * SN;
  float* G     = ws + 4 * SN;
  float* theta = G + SQ;
  float* svec  = theta + SQ;
  float* bvec  = svec + D_DIM;
  bf16_t* tT   = (bf16_t*)(bvec + D_DIM);  // 512KB packed bf16 theta (row-major)
  bf16_t* tC   = tT + SQ;                  // 512KB packed bf16 theta (transposed)

  // init state = (input, q, p); out slice 0 = input
  hipMemcpyAsync(state, x_in, SX * sizeof(float), hipMemcpyDeviceToDevice, stream);
  hipMemcpyAsync(state + SX, q_in, SQ * sizeof(float), hipMemcpyDeviceToDevice, stream);
  hipMemcpyAsync(state + SX + SQ, p_in, SQ * sizeof(float), hipMemcpyDeviceToDevice, stream);
  hipMemcpyAsync(out, x_in, SX * sizeof(float), hipMemcpyDeviceToDevice, stream);

  const dim3 blk(32, 8);
  const dim3 gDD(D_DIM / 64, D_DIM / 256);  // 512x512 outputs, MT=2
  const dim3 gMD(D_DIM / 64, M_DIM / 256);  // 8192x512 outputs, MT=2
  const dim3 gKD(D_DIM / 64, K_DIM / 256);  // 512x512 outputs (K rows), MT=2

  auto odefunc = [&](const float* s, float* k) {
    const float* sQ = s + SX;
    const float* sP = s + SX + SQ;
    // G = P^T @ relu(Q)
    gemm_wmma_kernel<true, false, 0, true, 0, 2>
        <<<gDD, blk, 0, stream>>>(sP, sQ, G, nullptr, nullptr, 1.0f, D_DIM);
    // theta = -(Mbar @ G)
    gemm_wmma_kernel<false, false, 0, false, 0, 2>
        <<<gDD, blk, 0, stream>>>(Mbar, G, theta, nullptr, nullptr, -1.0f, D_DIM);
    // pack theta -> bf16 (row-major + transposed), shared by the 3 big GEMMs
    pack_theta_kernel<<<dim3(16, 16), blk, 0, stream>>>(theta, tT, tC);
    // b = -(Mbar_b @ colsum(P))
    colsum_kernel<<<dim3(D_DIM / 256), dim3(256), 0, stream>>>(sP, svec);
    bvec_kernel<<<dim3(D_DIM / 256), dim3(256), 0, stream>>>(Mbarb, svec, bvec);
    // k.x = relu(x) @ theta^T + b      (dominant GEMM; B = packed bf16)
    gemm_wmma_kernel<false, true, 1, false, 1, 2>
        <<<gMD, blk, 0, stream>>>(s, tT, k, bvec, nullptr, 1.0f, M_DIM);
    // k.q = relu(Q) @ theta^T + b
    gemm_wmma_kernel<false, true, 1, false, 1, 2>
        <<<gKD, blk, 0, stream>>>(sQ, tT, k + SX, bvec, nullptr, 1.0f, K_DIM);
    // k.p = -( (Q>=0) * (P @ theta) )  (B = packed transposed bf16)
    gemm_wmma_kernel<false, false, 1, false, 2, 2>
        <<<gKD, blk, 0, stream>>>(sP, tC, k + SX + SQ, nullptr, sQ, -1.0f, K_DIM);
  };

  const int n4 = (int)(SN / 4);
  const int ewBlocks = (int)(SN / 4 / 256);  // SN is an exact multiple of 1024

  for (int iv = 0; iv < T_DIM - 1; ++iv) {
    for (int sub = 0; sub < N_SUBS; ++sub) {
      // k1
      odefunc(state, kbuf);
      ew_stage_kernel<<<ewBlocks, 256, 0, stream>>>(
          (float4*)tmp, (float4*)accb, (const float4*)state, (const float4*)kbuf,
          bt, iv, 0.5f, 0.0f, 1, n4);
      // k2
      odefunc(tmp, kbuf);
      ew_stage_kernel<<<ewBlocks, 256, 0, stream>>>(
          (float4*)tmp, (float4*)accb, (const float4*)state, (const float4*)kbuf,
          bt, iv, 0.5f, 2.0f, 0, n4);
      // k3
      odefunc(tmp, kbuf);
      ew_stage_kernel<<<ewBlocks, 256, 0, stream>>>(
          (float4*)tmp, (float4*)accb, (const float4*)state, (const float4*)kbuf,
          bt, iv, 1.0f, 2.0f, 0, n4);
      // k4 + combine
      odefunc(tmp, kbuf);
      ew_final_kernel<<<ewBlocks, 256, 0, stream>>>(
          (float4*)state, (const float4*)accb, (const float4*)kbuf, bt, iv, n4);
    }
    hipMemcpyAsync(out + (size_t)(iv + 1) * SX, state, SX * sizeof(float),
                   hipMemcpyDeviceToDevice, stream);
  }
}
